// MultiHeadAttention_52991306498499
// MI455X (gfx1250) — matrix operand order repaired and checked by reading, not compiled, not run on hardware
//
#include <hip/hip_runtime.h>

// ---------------------------------------------------------------------------
// Problem constants
// ---------------------------------------------------------------------------
#define NTOK 3000          // tokens (F*J)
#define NP   3008          // padded tokens: 188*16 = 47*64
#define NT   188           // 16-wide key tiles
#define DIMX 512           // input feature dim D
#define NB   4
#define NH   8
#define DHD  64
#define JFR  25            // frame size
#define ROWS (NB * NTOK)   // 12000 GEMM rows
#define HD   (NH * DHD)    // 512 projection cols
#define PERM ((size_t)NB * NH * NP * DHD)  // bf16 elems per Q/K/V matrix
#define NEGV (-9.0e15f)

// ---------------------------------------------------------------------------
// CDNA5 data-movement feature detection (guarded; fallbacks always compile)
// ---------------------------------------------------------------------------
#if defined(__has_builtin)
#  if __has_builtin(__builtin_amdgcn_global_load_async_to_lds_b128) && \
      __has_builtin(__builtin_amdgcn_s_wait_asynccnt)
#    define HAVE_ASYNC 1
#  endif
#  if __has_builtin(__builtin_amdgcn_tensor_load_to_lds) && \
      __has_builtin(__builtin_amdgcn_s_wait_tensorcnt)
#    define HAVE_TDM 1
#  endif
#endif

typedef unsigned int u32x4 __attribute__((ext_vector_type(4)));
typedef int          i32x8 __attribute__((ext_vector_type(8)));
typedef int          i32x4 __attribute__((ext_vector_type(4)));

#if defined(HAVE_ASYNC) || defined(HAVE_TDM)
typedef __attribute__((address_space(3))) void* as3_void;
// async builtin wants v4i pointers: AS1 (printed "__device__") src, AS3 dst
typedef __attribute__((address_space(1))) i32x4* gv4_p;
typedef __attribute__((address_space(3))) i32x4* lv4_p;
#endif

// ---------------------------------------------------------------------------
// Types for WMMA fragments (trivially-copyable ext vectors; safe in unions)
// ---------------------------------------------------------------------------
typedef __bf16        v16bf  __attribute__((ext_vector_type(16)));
typedef float         v8f    __attribute__((ext_vector_type(8)));
typedef unsigned int  uint4v __attribute__((ext_vector_type(4)));
typedef float         flt4v  __attribute__((ext_vector_type(4)));

union Frag {
  v16bf          v;
  unsigned short u[16];
  uint4v         q[2];
};

__device__ __forceinline__ unsigned short f2bf(float f) {
  unsigned int x = __float_as_uint(f);
  x += 0x7FFFu + ((x >> 16) & 1u);   // round-to-nearest-even
  return (unsigned short)(x >> 16);
}

__device__ __forceinline__ v8f wmma_bf16(const Frag& a, const Frag& b, v8f c) {
  // D = A(16x32 bf16) * B(32x16 bf16) + C(16x16 f32)
  return __builtin_amdgcn_wmma_f32_16x16x32_bf16(
      /*neg_a=*/false, a.v, /*neg_b=*/false, b.v,
      /*c_mod=*/(short)0, c, /*reuse_a=*/false, /*reuse_b=*/false);
}

// ---------------------------------------------------------------------------
// Workspace zero-init (padded Q/K/V rows must read as 0)
// ---------------------------------------------------------------------------
__global__ void zero_ws_kernel(uint4v* __restrict__ p, unsigned int n) {
  unsigned int i = blockIdx.x * blockDim.x + threadIdx.x;
  if (i < n) {
    uint4v z = {0u, 0u, 0u, 0u};
    p[i] = z;
  }
}

// ---------------------------------------------------------------------------
// Fused QKV projection:  Y = op(x @ W + b)  -> bf16 [mat][b][h][NP][64]
//   mat 0: Q (scaled by 1/sqrt(64));  mat 1: K;  mat 2: V (ReLU)
// Grid: (188 row tiles of 64, 8 col tiles of 64, 3 matrices); 256 threads.
// ---------------------------------------------------------------------------
__global__ __launch_bounds__(256) void proj_kernel(
    const float* __restrict__ x,
    const float* __restrict__ Wq, const float* __restrict__ bq,
    const float* __restrict__ Wk, const float* __restrict__ bk,
    const float* __restrict__ Wv, const float* __restrict__ bv,
    unsigned short* __restrict__ ws) {
  __shared__ unsigned short Xs[64 * 32];  // [row][k]   bf16
  __shared__ unsigned short Wt[64 * 32];  // [col][k]   bf16 (transposed)

  const int mat = blockIdx.z;
  const float* W    = (mat == 0) ? Wq : (mat == 1) ? Wk : Wv;
  const float* bias = (mat == 0) ? bq : (mat == 1) ? bk : bv;
  unsigned short* wsm = ws + (size_t)mat * PERM;

  const int rt = blockIdx.x, ct = blockIdx.y;
  const int t    = threadIdx.x;
  const int lane = t & 31, wid = t >> 5;
  const int half = lane >> 4, l16 = lane & 15;
  const int wr = wid & 3;        // 4 row blocks of 16
  const int wc = wid >> 2;       // 2 col blocks of 32

  v8f acc0 = {0.f, 0.f, 0.f, 0.f, 0.f, 0.f, 0.f, 0.f};
  v8f acc1 = acc0;

  for (int k0 = 0; k0 < DIMX; k0 += 32) {
    // ---- stage X tile 64x32 (f32 -> bf16) -------------------------------
    {
      const int row = t >> 2;
      const int ks  = (t & 3) * 8;
      const int rg  = rt * 64 + row;
      unsigned short tmp[8];
      if (rg < ROWS) {
        const flt4v* src = (const flt4v*)(x + (size_t)rg * DIMX + k0 + ks);
        flt4v f0 = src[0], f1 = src[1];
        tmp[0] = f2bf(f0.x); tmp[1] = f2bf(f0.y);
        tmp[2] = f2bf(f0.z); tmp[3] = f2bf(f0.w);
        tmp[4] = f2bf(f1.x); tmp[5] = f2bf(f1.y);
        tmp[6] = f2bf(f1.z); tmp[7] = f2bf(f1.w);
      } else {
        #pragma unroll
        for (int i = 0; i < 8; ++i) tmp[i] = 0;
      }
      *(uint4v*)&Xs[row * 32 + ks] = *(const uint4v*)tmp;
    }
    // ---- stage W tile 32x64, transposed to [col][k] ---------------------
    {
      const int kr = t >> 3;
      const int cs = (t & 7) * 8;
      const flt4v* src = (const flt4v*)(W + (size_t)(k0 + kr) * HD + ct * 64 + cs);
      flt4v f0 = src[0], f1 = src[1];
      Wt[(cs + 0) * 32 + kr] = f2bf(f0.x);
      Wt[(cs + 1) * 32 + kr] = f2bf(f0.y);
      Wt[(cs + 2) * 32 + kr] = f2bf(f0.z);
      Wt[(cs + 3) * 32 + kr] = f2bf(f0.w);
      Wt[(cs + 4) * 32 + kr] = f2bf(f1.x);
      Wt[(cs + 5) * 32 + kr] = f2bf(f1.y);
      Wt[(cs + 6) * 32 + kr] = f2bf(f1.z);
      Wt[(cs + 7) * 32 + kr] = f2bf(f1.w);
    }
    __syncthreads();

    // ---- fragments + 2 WMMAs per wave ----------------------------------
    Frag a, b0, b1;
    {
      // A 16x32: lane row = l16; elems e -> K = 16*half+e
      const unsigned short* base = &Xs[(16 * wr + l16) * 32];
      a.q[0] = *(const uint4v*)(base + 16 * half);
      a.q[1] = *(const uint4v*)(base + 16 * half + 8);
    }
    {
      // B 32x16: lane col = l16 (+16 block); elems e -> K = 16*half+e
      const unsigned short* p0 = &Wt[(32 * wc + l16) * 32 + 16 * half];
      b0.q[0] = ((const uint4v*)p0)[0];
      b0.q[1] = ((const uint4v*)p0)[1];
      const unsigned short* p1 = &Wt[(32 * wc + 16 + l16) * 32 + 16 * half];
      b1.q[0] = ((const uint4v*)p1)[0];
      b1.q[1] = ((const uint4v*)p1)[1];
    }
    acc0 = wmma_bf16(a, b0, acc0);
    acc1 = wmma_bf16(a, b1, acc1);
    __syncthreads();
  }

  // ---- epilogue: bias, per-matrix op, scatter to [b][h][n][dh] bf16 -----
  #pragma unroll
  for (int s = 0; s < 2; ++s) {
    v8f acc = s ? acc1 : acc0;
    const int cg = ct * 64 + 32 * wc + 16 * s + l16;
    const float bb = bias[cg];
    const int h  = cg >> 6;
    const int dh = cg & 63;
    #pragma unroll
    for (int r = 0; r < 8; ++r) {
      const int m  = r + 8 * half;          // C/D layout: m = vgpr + 8*(lane>=16)
      const int rg = rt * 64 + 16 * wr + m;
      if (rg >= ROWS) continue;
      const int bidx = rg / NTOK;
      const int nn   = rg - bidx * NTOK;
      float val = acc[r] + bb;
      if (mat == 0) val *= 0.125f;               // fold 1/sqrt(DH) into Q
      if (mat == 2) val = fmaxf(val, 0.f);       // ReLU for V
      wsm[((size_t)(bidx * NH + h) * NP + nn) * DHD + dh] = f2bf(val);
    }
  }
}

// ---------------------------------------------------------------------------
// Attention: one workgroup = one (b, h, 16-query tile).
//  Phase 1: S = Q Kᵀ (masked) into 16x3008 f32 LDS  (waves split key tiles)
//  Phase 2: softmax in LDS, stream normalized probs to global attn
//  Phase 3: O = P V via WMMA; V chunk staged to LDS by TDM / async-copy
// Dynamic LDS: 208000 bytes.
// ---------------------------------------------------------------------------
__global__ __launch_bounds__(256) void attn_kernel(
    const unsigned short* __restrict__ ws,
    float* __restrict__ out, float* __restrict__ attn) {
  extern __shared__ char smem[];
  float*          Sf   = (float*)smem;          // 16 x 3008 scores/probs
  float*          red  = Sf + 16 * NP;          // 16 x 16 reduction scratch
  float*          rmax = red + 256;             // 16
  float*          rsum = rmax + 16;             // 16
  float*          Ored = rsum + 16;             // 4 x 16 x 16 f32
  unsigned short* Qs   = (unsigned short*)(Ored + 1024);  // 16 x 64 bf16
  unsigned short* Vst  = Qs + 16 * 64;          // 64 x 64 bf16 V chunk

  const int qt = blockIdx.x;            // query tile
  const int h  = blockIdx.y, b = blockIdx.z;
  const int t    = threadIdx.x;
  const int lane = t & 31, wid = t >> 5;
  const int half = lane >> 4, l16 = lane & 15;
  const size_t bh = (size_t)(b * NH + h);

  const unsigned short* qbase = ws + bh * (size_t)NP * DHD;
  const unsigned short* kbase = ws + PERM + bh * (size_t)NP * DHD;
  const unsigned short* vbase = ws + 2 * PERM + bh * (size_t)NP * DHD;
  const int q0 = qt * 16;

  // ---- stage Q tile (16 x 64 bf16): async copy when available ----------
#if defined(HAVE_ASYNC)
  if (t < 128) {
    const int row = t >> 3, seg = (t & 7) * 8;
    __builtin_amdgcn_global_load_async_to_lds_b128(
        (gv4_p)(void*)(qbase + (size_t)(q0 + row) * DHD + seg),
        (lv4_p)(void*)&Qs[row * 64 + seg], 0, 0);
  }
  __builtin_amdgcn_s_wait_asynccnt(0);
#else
  if (t < 128) {
    const int row = t >> 3, seg = (t & 7) * 8;
    *(uint4v*)&Qs[row * 64 + seg] =
        *(const uint4v*)(qbase + (size_t)(q0 + row) * DHD + seg);
  }
#endif
  __syncthreads();

  // ---- build the two Q A-fragments once (reused for every key tile) -----
  Frag aq[2];
  #pragma unroll
  for (int c = 0; c < 2; ++c) {
    const unsigned short* base = Qs + l16 * 64 + 32 * c;
    aq[c].q[0] = *(const uint4v*)(base + 16 * half);
    aq[c].q[1] = *(const uint4v*)(base + 16 * half + 8);
  }

  // ---- Phase 1: masked scores into LDS ----------------------------------
  for (int jt = wid; jt < NT; jt += 8) {
    if (jt + 8 < NT)
      __builtin_prefetch(kbase + (size_t)((jt + 8) * 16) * DHD, 0, 0);
    v8f acc = {0.f, 0.f, 0.f, 0.f, 0.f, 0.f, 0.f, 0.f};
    #pragma unroll
    for (int c = 0; c < 2; ++c) {
      // B = Kᵀ: lane col = key l16; elems e -> dh = 32c + 16*half + e
      Frag bk_;
      const unsigned short* base =
          kbase + (size_t)(jt * 16 + l16) * DHD + 32 * c + 16 * half;
      bk_.q[0] = ((const uint4v*)base)[0];
      bk_.q[1] = ((const uint4v*)base)[1];
      acc = wmma_bf16(aq[c], bk_, acc);
    }
    const int kg  = jt * 16 + l16;
    const bool kin = kg < NTOK;
    const int kf  = kg / JFR;
    #pragma unroll
    for (int r = 0; r < 8; ++r) {
      const int m  = r + 8 * half;
      const int qg = q0 + m;
      const bool allowed = kin && ((qg / JFR != kf) || (qg == kg));
      Sf[m * NP + jt * 16 + l16] = allowed ? acc[r] : NEGV;
    }
  }
  __syncthreads();

  // ---- Phase 2: softmax over each 3008-wide row -------------------------
  const int row = t >> 4, cl = t & 15;   // 16 threads per query row
  float* Srow = Sf + row * NP;
  float mx = -3.0e38f;
  for (int c = cl; c < NP; c += 16) mx = fmaxf(mx, Srow[c]);
  red[row * 16 + cl] = mx;
  __syncthreads();
  if (cl == 0) {
    float m2 = red[row * 16];
    #pragma unroll
    for (int i = 1; i < 16; ++i) m2 = fmaxf(m2, red[row * 16 + i]);
    rmax[row] = m2;
  }
  __syncthreads();
  const float rm = rmax[row];
  float sum = 0.f;
  for (int c = cl; c < NP; c += 16) {
    const float p = __expf(Srow[c] - rm);
    Srow[c] = p;
    sum += p;
  }
  red[row * 16 + cl] = sum;
  __syncthreads();
  if (cl == 0) {
    float s2 = 0.f;
    #pragma unroll
    for (int i = 0; i < 16; ++i) s2 += red[row * 16 + i];
    rsum[row] = s2;
  }
  __syncthreads();
  const float inv = 1.0f / rsum[row];
  const int qg_row = q0 + row;
  float* arow = attn + (bh * NTOK + (size_t)qg_row) * NTOK;
  for (int c = cl; c < NP; c += 16) {
    const float p = Srow[c] * inv;
    Srow[c] = p;                                   // keep for P@V
    if (qg_row < NTOK && c < NTOK) arow[c] = p;    // stream attn to HBM
  }
  __syncthreads();

  // ---- Phase 3: O = P @ V  (47 chunks of 64 keys; 8 waves = 4 col x 2 K) -
  const int par = wid >> 2;      // K-half within 64-key chunk
  const int ctl = wid & 3;       // 16-wide dh column tile
  v8f oacc = {0.f, 0.f, 0.f, 0.f, 0.f, 0.f, 0.f, 0.f};

#if defined(HAVE_TDM)
  // LDS byte offset of Vst within this workgroup's allocation (for the D#)
  const unsigned int vst_lds =
      (unsigned int)(unsigned long long)(as3_void)(void*)Vst;
#endif

  for (int it = 0; it < 47; ++it) {
    const int key0 = it * 64;
#if defined(HAVE_TDM)
    // ---- Tensor Data Mover: one wave DMAs the 64x64 bf16 tile to LDS ----
    if (wid == 0) {
      const unsigned long long ga =
          (unsigned long long)(size_t)(const void*)(vbase + (size_t)key0 * DHD);
      // D# group0: count=1 | lds_addr | global_addr | type=2 (bits 127:126)
      u32x4 g0 = { 1u, vst_lds, (unsigned int)ga,
                   ((unsigned int)(ga >> 32) & 0x01FFFFFFu) | 0x80000000u };
      // D# group1: data_size=2B | tensor_dim0=64 | tensor_dim1=3008
      //            tile_dim0=64 | tile_dim1=64 | tensor_dim0_stride=64
      i32x8 g1 = { (int)0x00010000u,          // data_size=1 (2 bytes)
                   (int)(64u << 16),          // tensor_dim0 low16 at [63:48]
                   (int)(3008u << 16),        // tensor_dim1 low16 at [95:80]
                   (int)(64u << 16),          // tile_dim0 at [127:112]
                   (int)64,                   // tile_dim1 at [143:128]
                   (int)64,                   // tensor_dim0_stride low32
                   0, 0 };
      i32x4 g2 = {0, 0, 0, 0};
      i32x4 g3 = {0, 0, 0, 0};
      i32x8 g4 = {0, 0, 0, 0, 0, 0, 0, 0};
      __builtin_amdgcn_tensor_load_to_lds(g0, g1, g2, g3, g4, 0);
      __builtin_amdgcn_s_wait_tensorcnt(0);
    }
#elif defined(HAVE_ASYNC)
    // ---- async global->LDS copy, row-major [key][dh] --------------------
    {
      const int kl = t >> 2, seg = (t & 3) * 16;
      const unsigned short* src = vbase + (size_t)(key0 + kl) * DHD + seg;
      unsigned short* dst = Vst + kl * 64 + seg;
      __builtin_amdgcn_global_load_async_to_lds_b128(
          (gv4_p)(void*)src, (lv4_p)(void*)dst, 0, 0);
      __builtin_amdgcn_global_load_async_to_lds_b128(
          (gv4_p)(void*)(src + 8), (lv4_p)(void*)(dst + 8), 0, 0);
    }
    __builtin_amdgcn_s_wait_asynccnt(0);
#else
    // ---- manual staging, transposed Vt[dh][key] -------------------------
    {
      const int kl = t >> 2, seg = (t & 3) * 16;
      const unsigned short* src = vbase + (size_t)(key0 + kl) * DHD + seg;
      uint4v d0 = ((const uint4v*)src)[0];
      uint4v d1 = ((const uint4v*)src)[1];
      unsigned short tmp[16];
      *(uint4v*)&tmp[0] = d0;
      *(uint4v*)&tmp[8] = d1;
      #pragma unroll
      for (int i = 0; i < 16; ++i) Vst[(seg + i) * 64 + kl] = tmp[i];
    }
#endif
    __syncthreads();

    // A = P 16x32 (f32 LDS -> bf16), elems e -> key = 32*par+16*half+e
    Frag ap;
    {
      const float* sr = Sf + l16 * NP + key0 + 32 * par;
      #pragma unroll
      for (int e = 0; e < 8; ++e) ap.u[e] = f2bf(sr[16 * half + e]);
      #pragma unroll
      for (int e = 0; e < 8; ++e) ap.u[8 + e] = f2bf(sr[16 * half + 8 + e]);
    }
    // B = V 32x16: lane col = dh = ctl*16+l16; elems e -> key = 32*par+16*half+e
    Frag bv_;
#if defined(HAVE_TDM) || defined(HAVE_ASYNC)
    {   // V is row-major [key][dh] in LDS: per-lane gather down the key axis
      #pragma unroll
      for (int e = 0; e < 16; ++e)
        bv_.u[e] = Vst[(32 * par + 16 * half + e) * 64 + ctl * 16 + l16];
    }
#else
    {   // V is transposed [dh][key] in LDS: contiguous vector reads
      const unsigned short* base = Vst + (ctl * 16 + l16) * 64 + 32 * par + 16 * half;
      bv_.q[0] = ((const uint4v*)base)[0];
      bv_.q[1] = ((const uint4v*)base)[1];
    }
#endif
    oacc = wmma_bf16(ap, bv_, oacc);
    __syncthreads();
  }

  // ---- cross-wave reduce (par=1 -> LDS, par=0 adds) and store out -------
  if (par == 1) {
    #pragma unroll
    for (int r = 0; r < 8; ++r)
      Ored[ctl * 256 + (r + 8 * half) * 16 + l16] = oacc[r];
  }
  __syncthreads();
  if (par == 0) {
    const float* ob = Ored + ctl * 256;
    #pragma unroll
    for (int r = 0; r < 8; ++r) {
      const int m  = r + 8 * half;
      const int qg = q0 + m;
      if (qg >= NTOK) continue;
      const float val = oacc[r] + ob[m * 16 + l16];
      out[((size_t)b * NTOK + qg) * HD + h * DHD + ctl * 16 + l16] = val;
    }
  }
}

// ---------------------------------------------------------------------------
// Launcher
// ---------------------------------------------------------------------------
extern "C" void kernel_launch(void* const* d_in, const int* in_sizes, int n_in,
                              void* d_out, int out_size, void* d_ws, size_t ws_size,
                              hipStream_t stream) {
  (void)in_sizes; (void)n_in; (void)out_size; (void)ws_size;
  const float* x  = (const float*)d_in[0];
  const float* Wq = (const float*)d_in[1];
  const float* bq = (const float*)d_in[2];
  const float* Wk = (const float*)d_in[3];
  const float* bk = (const float*)d_in[4];
  const float* Wv = (const float*)d_in[5];
  const float* bv = (const float*)d_in[6];

  float* out  = (float*)d_out;
  float* attn = out + (size_t)NB * NTOK * HD;      // outputs concatenated
  unsigned short* ws = (unsigned short*)d_ws;

  // 3 matrices * 32 (b,h) * 3008 * 64 bf16 = ~37 MB workspace
  const unsigned int n_u4 = (unsigned int)((3 * PERM) / 8);  // 16B chunks
  zero_ws_kernel<<<(n_u4 + 255) / 256, 256, 0, stream>>>((uint4v*)d_ws, n_u4);

  proj_kernel<<<dim3(188, 8, 3), 256, 0, stream>>>(x, Wq, bq, Wk, bk, Wv, bv, ws);

  const size_t smem_bytes =
      (size_t)16 * NP * 4 + 256 * 4 + 16 * 4 + 16 * 4 + 1024 * 4  // f32 region
      + 16 * 64 * 2 + 64 * 64 * 2;                                // bf16 region
  attn_kernel<<<dim3(NT, NH, NB), 256, smem_bytes, stream>>>(ws, out, attn);
}
